// SpectralAttentionLayer_21311627723298
// MI455X (gfx1250) — matrix operand order repaired and checked by reading, not compiled, not run on hardware
//
#include <hip/hip_runtime.h>

typedef __attribute__((ext_vector_type(16))) _Float16 v16h;
typedef __attribute__((ext_vector_type(8)))  float    v8f;

#define DF 32
#define NEG_SLOPE 0.2f

static __host__ __device__ inline int cdiv(int a, int b) { return (a + b - 1) / b; }

// ---------------------------------------------------------------------------
// init: zero agg + out, zero deg/den, mx = -inf
// ---------------------------------------------------------------------------
__global__ void k_init(float* __restrict__ agg, float* __restrict__ out,
                       float* __restrict__ deg, float* __restrict__ den,
                       float* __restrict__ mx, int N) {
  int i = blockIdx.x * blockDim.x + threadIdx.x;
  if (i < N * DF) { agg[i] = 0.f; out[i] = 0.f; }
  if (i < N) { deg[i] = 0.f; den[i] = 0.f; mx[i] = -__builtin_inff(); }
}

__global__ void k_zero(float* __restrict__ p, int n) {
  int i = blockIdx.x * blockDim.x + threadIdx.x;
  if (i < n) p[i] = 0.f;
}

// ---------------------------------------------------------------------------
// degree + Dinv
// ---------------------------------------------------------------------------
__global__ void k_deg(const int* __restrict__ dst, float* __restrict__ deg, int E) {
  int e = blockIdx.x * blockDim.x + threadIdx.x;
  if (e < E) atomicAdd(&deg[dst[e]], 1.0f);
}

__global__ void k_dinv(float* __restrict__ deg, int N) {
  int i = blockIdx.x * blockDim.x + threadIdx.x;
  if (i < N) {
    float d = deg[i];
    d = d < 1.0f ? 1.0f : d;
    deg[i] = 1.0f / sqrtf(d);   // becomes Dinv in place
  }
}

// ---------------------------------------------------------------------------
// unnL scatter:  agg[dst] += x[src] * Dinv[src]   (32 lanes per edge)
// ---------------------------------------------------------------------------
__global__ void k_scatter(const float* __restrict__ x, const float* __restrict__ Dinv,
                          const int* __restrict__ src, const int* __restrict__ dst,
                          float* __restrict__ agg, int E) {
  int t = blockIdx.x * blockDim.x + threadIdx.x;
  int e = t >> 5, d = t & 31;
  if (e >= E) return;
  int s = src[e], q = dst[e];
  float v = x[s * DF + d] * Dinv[s];
  atomicAdd(&agg[q * DF + d], v);
}

// ---------------------------------------------------------------------------
// Chebyshev combines
// ---------------------------------------------------------------------------
__global__ void k_combine1(const float* __restrict__ u, const float* __restrict__ agg,
                           const float* __restrict__ Dinv, const float* __restrict__ lam,
                           float* __restrict__ X1, int N) {
  int i = blockIdx.x * blockDim.x + threadIdx.x;
  if (i >= N * DF) return;
  float rn = 2.0f / lam[0];
  int node = i >> 5;
  float h = Dinv[node] * agg[i];
  X1[i] = -rn * h + u[i] * (rn - 1.0f);
}

// X2 written in place over agg (elementwise safe)
__global__ void k_combine2(const float* __restrict__ u, const float* __restrict__ X1,
                           float* __restrict__ agg, const float* __restrict__ Dinv,
                           const float* __restrict__ lam, int N) {
  int i = blockIdx.x * blockDim.x + threadIdx.x;
  if (i >= N * DF) return;
  float rn = 2.0f / lam[0];
  int node = i >> 5;
  float h = Dinv[node] * agg[i];
  agg[i] = -2.0f * rn * h + X1[i] * (2.0f * (rn - 1.0f)) - u[i];
}

// ---------------------------------------------------------------------------
// Repack f32 weights into per-lane WMMA B-matrix f16 layout.
// B (32x16 f16): lane<16 -> n=lane,  k = i (i<8) else 16+i-8
//                lane>=16 -> n=lane-16, k = 8+i (i<8) else 24+i-8
// Storage: B16[(tile*32 + lane)*16 + i]
// ---------------------------------------------------------------------------
__global__ void k_repack_cheb(const float* __restrict__ W, _Float16* __restrict__ B16) {
  int idx = blockIdx.x * blockDim.x + threadIdx.x;            // 6*32*16 = 3072
  if (idx >= 6 * 32 * 16) return;
  int i = idx & 15, lane = (idx >> 4) & 31, tc = idx >> 9;    // tc = c*2 + ntile
  int c = tc >> 1, ntile = tc & 1;
  int klocal = ((lane >> 4) << 3) + (i & 7) + ((i >> 3) << 4);
  int n = (lane & 15) + 16 * ntile;
  B16[idx] = (_Float16)W[(32 * c + klocal) * DF + n];
}

__global__ void k_repack_att(const float* __restrict__ srcW, const float* __restrict__ dstW,
                             _Float16* __restrict__ B16) {
  int idx = blockIdx.x * blockDim.x + threadIdx.x;            // 4*32*16 = 2048
  if (idx >= 4 * 32 * 16) return;
  int i = idx & 15, lane = (idx >> 4) & 31, tc = idx >> 9;    // tc 0..3: [src0,src1,dst0,dst1]
  const float* W = (tc < 2) ? srcW : dstW;
  int ntile = tc & 1;
  int klocal = ((lane >> 4) << 3) + (i & 7) + ((i >> 3) << 4);
  int n = (lane & 15) + 16 * ntile;
  B16[idx] = (_Float16)W[klocal * DF + n];
}

// ---------------------------------------------------------------------------
// WMMA A-tile loader: 16-bit A 16x32 layout.
//   m = lane&15, kb = (lane>>4)*8, element i: k = kb+i (i<8) else 16+kb+i-8
// ---------------------------------------------------------------------------
static __device__ __forceinline__ v16h load_a_tile(const float* __restrict__ row, int kb) {
  v16h a;
#pragma unroll
  for (int i = 0; i < 16; ++i) {
    int k = (i < 8) ? (kb + i) : (16 + kb + (i - 8));
    a[i] = (_Float16)row[k];
  }
  return a;
}

static __device__ __forceinline__ v16h load_b_tile(const _Float16* __restrict__ Bp) {
  v16h b;
#pragma unroll
  for (int i = 0; i < 16; ++i) b[i] = Bp[i];
  return b;
}

// Store one 16x16 f32 C/D tile (wave-uniform guard: no per-element EXEC churn
// on the full-tile fast path; guarded path only for a true tail group).
static __device__ __forceinline__ void store_tile(float* __restrict__ O, int base, int col,
                                                  int lane, const v8f& acc, float bv,
                                                  bool relu, int N) {
  int mrow0 = base + ((lane >> 4) << 3);
  if (base + 16 <= N) {
#pragma unroll
    for (int r = 0; r < 8; ++r) {
      float v = acc[r] + bv;
      if (relu) v = v > 0.f ? v : 0.f;
      O[(long)(mrow0 + r) * DF + col] = v;
    }
  } else {
#pragma unroll
    for (int r = 0; r < 8; ++r) {
      int orow = mrow0 + r;
      if (orow < N) {
        float v = acc[r] + bv;
        if (relu) v = v > 0.f ? v : 0.f;
        O[(long)orow * DF + col] = v;
      }
    }
  }
}

// ---------------------------------------------------------------------------
// GEMM1: hc = relu([u | X1 | X2] @ cheb_W + cheb_b), wave per 16 rows
// ---------------------------------------------------------------------------
__global__ void k_gemm_cheb(const float* __restrict__ X0, const float* __restrict__ X1,
                            const float* __restrict__ X2, const _Float16* __restrict__ chebB,
                            const float* __restrict__ bias, float* __restrict__ hc,
                            int N, int ngroups) {
  int wave = threadIdx.x >> 5;
  int lane = threadIdx.x & 31;
  int group = blockIdx.x * (blockDim.x >> 5) + wave;
  if (group >= ngroups) return;                 // wave-uniform: EXEC stays all-1s
  int base = group * 16;
  int m = lane & 15;
  int kb = (lane >> 4) << 3;
  int row = base + m; if (row >= N) row = N - 1;

  v16h a0 = load_a_tile(X0 + (long)row * DF, kb);
  v16h a1 = load_a_tile(X1 + (long)row * DF, kb);
  v16h a2 = load_a_tile(X2 + (long)row * DF, kb);

#pragma unroll
  for (int nt = 0; nt < 2; ++nt) {
    v8f acc = {};
    v16h b0 = load_b_tile(chebB + ((0 * 2 + nt) * 32 + lane) * 16);
    acc = __builtin_amdgcn_wmma_f32_16x16x32_f16(false, a0, false, b0, (short)0, acc, false, false);
    v16h b1 = load_b_tile(chebB + ((1 * 2 + nt) * 32 + lane) * 16);
    acc = __builtin_amdgcn_wmma_f32_16x16x32_f16(false, a1, false, b1, (short)0, acc, false, false);
    v16h b2 = load_b_tile(chebB + ((2 * 2 + nt) * 32 + lane) * 16);
    acc = __builtin_amdgcn_wmma_f32_16x16x32_f16(false, a2, false, b2, (short)0, acc, false, false);

    int col = (lane & 15) + 16 * nt;
    store_tile(hc, base, col, lane, acc, bias[col], /*relu=*/true, N);
  }
}

// ---------------------------------------------------------------------------
// GEMM2: fsrc = hc @ src_W + src_b ; fdst = hc @ dst_W + dst_b
// ---------------------------------------------------------------------------
__global__ void k_gemm_att(const float* __restrict__ hc, const _Float16* __restrict__ attB,
                           const float* __restrict__ srcb, const float* __restrict__ dstb,
                           float* __restrict__ fsrc, float* __restrict__ fdst,
                           int N, int ngroups) {
  int wave = threadIdx.x >> 5;
  int lane = threadIdx.x & 31;
  int group = blockIdx.x * (blockDim.x >> 5) + wave;
  if (group >= ngroups) return;
  int base = group * 16;
  int m = lane & 15;
  int kb = (lane >> 4) << 3;
  int row = base + m; if (row >= N) row = N - 1;

  v16h a = load_a_tile(hc + (long)row * DF, kb);

#pragma unroll
  for (int t = 0; t < 4; ++t) {
    v8f acc = {};
    v16h b = load_b_tile(attB + (t * 32 + lane) * 16);
    acc = __builtin_amdgcn_wmma_f32_16x16x32_f16(false, a, false, b, (short)0, acc, false, false);

    float* O = (t < 2) ? fsrc : fdst;
    const float* bb = (t < 2) ? srcb : dstb;
    int col = (lane & 15) + 16 * (t & 1);
    store_tile(O, base, col, lane, acc, bb[col], /*relu=*/false, N);
  }
}

// ---------------------------------------------------------------------------
// edge logits + segment max (global_atomic_max_num_f32 via __hip_atomic_fetch_max)
// ---------------------------------------------------------------------------
__global__ void k_logits(const float* __restrict__ fsrc, const float* __restrict__ fdst,
                         const float* __restrict__ attn, const int* __restrict__ src,
                         const int* __restrict__ dst, float* __restrict__ lg,
                         float* __restrict__ mx, int E) {
  int e = blockIdx.x * blockDim.x + threadIdx.x;
  if (e >= E) return;
  int s = src[e], q = dst[e];
  const float4* fs = (const float4*)(fsrc + (long)s * DF);
  const float4* fd = (const float4*)(fdst + (long)q * DF);
  const float4* at = (const float4*)attn;
  float sum = 0.f;
#pragma unroll
  for (int j = 0; j < 8; ++j) {
    float4 a = fs[j], b = fd[j], w = at[j];
    float x;
    x = a.x + b.x; x = x > 0.f ? x : NEG_SLOPE * x; sum += w.x * x;
    x = a.y + b.y; x = x > 0.f ? x : NEG_SLOPE * x; sum += w.y * x;
    x = a.z + b.z; x = x > 0.f ? x : NEG_SLOPE * x; sum += w.z * x;
    x = a.w + b.w; x = x > 0.f ? x : NEG_SLOPE * x; sum += w.w * x;
  }
  lg[e] = sum;
  (void)__hip_atomic_fetch_max(&mx[q], sum, __ATOMIC_RELAXED, __HIP_MEMORY_SCOPE_AGENT);
}

// ex = exp(logit - mx[dst]) (in place over lg) ; den[dst] += ex
__global__ void k_expden(float* __restrict__ lg, const float* __restrict__ mx,
                         const int* __restrict__ dst, float* __restrict__ den, int E) {
  int e = blockIdx.x * blockDim.x + threadIdx.x;
  if (e >= E) return;
  int q = dst[e];
  float v = expf(lg[e] - mx[q]);
  lg[e] = v;
  atomicAdd(&den[q], v);
}

// out[dst] += fsrc[src] * (ex / den[dst])
__global__ void k_final(const float* __restrict__ fsrc, const float* __restrict__ lg,
                        const float* __restrict__ den, const int* __restrict__ src,
                        const int* __restrict__ dst, float* __restrict__ out, int E) {
  int t = blockIdx.x * blockDim.x + threadIdx.x;
  int e = t >> 5, d = t & 31;
  if (e >= E) return;
  int s = src[e], q = dst[e];
  float dn = den[q];
  dn = dn > 0.f ? dn : 1.0f;
  float a = lg[e] / dn;
  atomicAdd(&out[q * DF + d], fsrc[s * DF + d] * a);
}

// ---------------------------------------------------------------------------
extern "C" void kernel_launch(void* const* d_in, const int* in_sizes, int n_in,
                              void* d_out, int out_size, void* d_ws, size_t ws_size,
                              hipStream_t stream) {
  const float* u     = (const float*)d_in[0];
  const float* lam   = (const float*)d_in[1];
  const int*   esrc  = (const int*)d_in[2];
  const int*   edst  = (const int*)d_in[3];
  const float* chebW = (const float*)d_in[4];
  const float* chebb = (const float*)d_in[5];
  const float* srcW  = (const float*)d_in[6];
  const float* srcb  = (const float*)d_in[7];
  const float* dstW  = (const float*)d_in[8];
  const float* dstb  = (const float*)d_in[9];
  const float* attn  = (const float*)d_in[10];

  const int N = in_sizes[0] / DF;
  const int E = in_sizes[2];

  // workspace layout (floats)
  float* ws   = (float*)d_ws;
  float* Dinv = ws;                        // N   (deg -> Dinv in place)
  float* agg  = Dinv + N;                  // N*32 (later reused as X2)
  float* X1   = agg  + (long)N * DF;       // N*32
  float* hc   = X1   + (long)N * DF;       // N*32
  float* fsrc = hc   + (long)N * DF;       // N*32
  float* fdst = fsrc + (long)N * DF;       // N*32
  float* mx   = fdst + (long)N * DF;       // N
  float* den  = mx + N;                    // N
  float* lg   = den + N;                   // E (logits -> ex in place)
  _Float16* chebB = (_Float16*)(lg + E);   // 6*32*16 halves
  _Float16* attB  = chebB + 6 * 32 * 16;   // 4*32*16 halves

  float* out = (float*)d_out;

  const int B = 256;
  const int ngroups = cdiv(N, 16);
  const int gw = cdiv(ngroups, B / 32);

  // 0) init accumulators
  k_init<<<cdiv(N * DF, B), B, 0, stream>>>(agg, out, Dinv, den, mx, N);
  // 1) weight repack (f32 -> f16 WMMA B layout)
  k_repack_cheb<<<cdiv(6 * 32 * 16, B), B, 0, stream>>>(chebW, chebB);
  k_repack_att<<<cdiv(4 * 32 * 16, B), B, 0, stream>>>(srcW, dstW, attB);
  // 2) degrees -> Dinv
  k_deg<<<cdiv(E, B), B, 0, stream>>>(edst, Dinv, E);
  k_dinv<<<cdiv(N, B), B, 0, stream>>>(Dinv, N);
  // 3) Chebyshev: X1
  k_scatter<<<cdiv(E * DF, B), B, 0, stream>>>(u, Dinv, esrc, edst, agg, E);
  k_combine1<<<cdiv(N * DF, B), B, 0, stream>>>(u, agg, Dinv, lam, X1, N);
  // 4) Chebyshev: X2 (agg reused, then overwritten in place by X2)
  k_zero<<<cdiv(N * DF, B), B, 0, stream>>>(agg, N * DF);
  k_scatter<<<cdiv(E * DF, B), B, 0, stream>>>(X1, Dinv, esrc, edst, agg, E);
  k_combine2<<<cdiv(N * DF, B), B, 0, stream>>>(u, X1, agg, Dinv, lam, N);
  // 5) WMMA GEMMs
  k_gemm_cheb<<<gw, B, 0, stream>>>(u, X1, /*X2=*/agg, chebB, chebb, hc, N, ngroups);
  k_gemm_att<<<gw, B, 0, stream>>>(hc, attB, srcb, dstb, fsrc, fdst, N, ngroups);
  // 6) edge softmax + aggregation
  k_logits<<<cdiv(E, B), B, 0, stream>>>(fsrc, fdst, attn, esrc, edst, lg, mx, E);
  k_expden<<<cdiv(E, B), B, 0, stream>>>(lg, mx, edst, den, E);
  k_final<<<cdiv(E * DF, B), B, 0, stream>>>(fsrc, lg, den, esrc, edst, out, E);
}
